// MultiHeadAttention_82480551953118
// MI455X (gfx1250) — matrix operand order repaired and checked by reading, not compiled, not run on hardware
//
#include <hip/hip_runtime.h>
#include <hip/hip_bf16.h>

// MI455X / gfx1250, wave32, WMMA f32_16x16x32_f16.
// Fused MHA: cvt -> QKV proj GEMMs -> flash attention (async-LDS double
// buffered K/V staging) -> out proj.

typedef __attribute__((ext_vector_type(16))) _Float16 v16h;
typedef __attribute__((ext_vector_type(8)))  _Float16 half8;
typedef __attribute__((ext_vector_type(4)))  _Float16 half4;
typedef __attribute__((ext_vector_type(8)))  float    v8f;
typedef int v4i_ __attribute__((vector_size(16)));   // builtin's pointee type

#define D_MODEL   512
#define NUM_HEADS 8
#define D_HEAD    64
#define BATCH     4
#define SEQ       2048
#define M_TOTAL   (BATCH*SEQ)   // 8192 rows

#define WMMA(a,b,c) __builtin_amdgcn_wmma_f32_16x16x32_f16(false,(a),false,(b),(short)0,(c),false,false)

// gfx1250 async global->LDS staging (ASYNCcnt path), with sync fallback.
#if __has_builtin(__builtin_amdgcn_global_load_async_to_lds_b128) && \
    __has_builtin(__builtin_amdgcn_s_wait_asynccnt)
#define ASYNC_STAGE 1
#define ASYNC_CP16(dst, src) \
  __builtin_amdgcn_global_load_async_to_lds_b128( \
      (__attribute__((address_space(1))) v4i_*)(src), \
      (__attribute__((address_space(3))) v4i_*)(dst), 0, 0)
#define WAIT_ASYNC(n) __builtin_amdgcn_s_wait_asynccnt(n)
#else
#define ASYNC_STAGE 0
#define ASYNC_CP16(dst, src) (*(half8*)(dst) = *(const half8*)(src))
#define WAIT_ASYNC(n)
#endif

// A-fragment 16x32 f16 (ISA 7.12.2): lane holds row m=lane&15; half=lane>>4
// selects k-chunks [half*8, half*8+8) and [16+half*8, ...). Two b128 loads.
static __device__ __forceinline__ v16h frag_a(const _Float16* base, int stride, int kstep, int lane){
  const int row = lane & 15, h = lane >> 4;
  const _Float16* p = base + (size_t)row*stride + kstep + h*8;
  union { v16h v; half8 q[2]; } u;
  u.q[0] = *(const half8*)(p);
  u.q[1] = *(const half8*)(p + 16);
  return u.v;
}
// B-fragment 32x16 f16: lane holds column n=lane&15; half selects k-chunks
// [half*8, half*8+8) and [16+half*8, ...), read from K-contiguous storage (stride between columns).
static __device__ __forceinline__ v16h frag_b(const _Float16* base, int stride, int kstep, int lane){
  const int n = lane & 15, h = lane >> 4;
  const _Float16* p = base + (size_t)n*stride + kstep + h*8;
  union { v16h v; half8 q[2]; } u;
  u.q[0] = *(const half8*)(p);
  u.q[1] = *(const half8*)(p + 16);
  return u.v;
}

// ---------------- fp32 -> f16 conversions ----------------
__global__ void cvt_x_kernel(const float* __restrict__ x, _Float16* __restrict__ xh, int n){
  int i = (blockIdx.x*blockDim.x + threadIdx.x)*4;
  if (i < n){
    const float4 f = *(const float4*)(x + i);
    half4 h = { (_Float16)f.x, (_Float16)f.y, (_Float16)f.z, (_Float16)f.w };
    *(half4*)(xh + i) = h;
  }
}

__global__ void cvt_w_kernel(const float* __restrict__ Wq, const float* __restrict__ Wk,
                             const float* __restrict__ Wv, const float* __restrict__ Wo,
                             _Float16* __restrict__ WT){
  int i = blockIdx.x*blockDim.x + threadIdx.x;      // 0..262143
  const float* W = (blockIdx.y==0)?Wq:(blockIdx.y==1)?Wk:(blockIdx.y==2)?Wv:Wo;
  int k = i >> 9, n = i & 511;                      // W is [K][N]; store WT[N][K]
  WT[(size_t)blockIdx.y*D_MODEL*D_MODEL + (size_t)n*D_MODEL + k] = (_Float16)W[i];
}

// ---------------- QKV projection GEMM ----------------
// C[8192x512] = xh[8192x512] @ W[512x512] + b ; scatter-store into head layout.
// mode 0/1: dst[bh][s][64] (Q,K)   mode 2: dst[bh][d][s] (V transposed)
__global__ __launch_bounds__(256) void proj_kernel(const _Float16* __restrict__ A,
    const _Float16* __restrict__ WT, const float* __restrict__ bias,
    _Float16* __restrict__ dst, int mode)
{
  const int lane = threadIdx.x & 31;
  const int w    = threadIdx.x >> 5;
  const int m0 = blockIdx.x*128 + (w & 3)*32;
  const int n0 = blockIdx.y*64  + (w >> 2)*32;
  v8f c[2][2] = {};
  for (int k = 0; k < D_MODEL; k += 32){
    __builtin_prefetch(A  + (size_t)m0*D_MODEL + k + 64, 0, 1);
    __builtin_prefetch(WT + (size_t)n0*D_MODEL + k + 64, 0, 1);
    v16h a0 = frag_a(A  + (size_t)m0*D_MODEL,        D_MODEL, k, lane);
    v16h a1 = frag_a(A  + (size_t)(m0+16)*D_MODEL,   D_MODEL, k, lane);
    v16h b0 = frag_b(WT + (size_t)n0*D_MODEL,        D_MODEL, k, lane);
    v16h b1 = frag_b(WT + (size_t)(n0+16)*D_MODEL,   D_MODEL, k, lane);
    c[0][0] = WMMA(a0, b0, c[0][0]);
    c[0][1] = WMMA(a0, b1, c[0][1]);
    c[1][0] = WMMA(a1, b0, c[1][0]);
    c[1][1] = WMMA(a1, b1, c[1][1]);
  }
  const int half = lane >> 4, nn = lane & 15;
  #pragma unroll
  for (int tm = 0; tm < 2; ++tm)
  #pragma unroll
  for (int tn = 0; tn < 2; ++tn)
  #pragma unroll
  for (int r = 0; r < 8; ++r){
    int row = m0 + tm*16 + r + half*8;
    int col = n0 + tn*16 + nn;
    _Float16 v = (_Float16)(c[tm][tn][r] + bias[col]);
    int bb = row >> 11, sp = row & 2047;      // batch, seq pos
    int hh = col >> 6,  dd = col & 63;        // head, head dim
    int bh = bb*NUM_HEADS + hh;
    if (mode < 2)
      dst[((size_t)bh*SEQ + sp)*D_HEAD + dd] = v;
    else
      dst[((size_t)bh*D_HEAD + dd)*SEQ + sp] = v;
  }
}

// ---------------- flash attention ----------------
// Grid (S/128, B*H); 8 waves/WG, 16 queries/wave, key blocks of 32.
// K/V tiles are staged into LDS once per WG with async global->LDS copies,
// double-buffered so block i+1's DMA overlaps block i's WMMA compute.
#define KS_STRIDE 72   // f16; 144B: 16B-aligned rows, conflict-free banks
#define VS_STRIDE 40   // f16;  80B: 16B-aligned rows, conflict-free banks
#define NKB (SEQ/32)   // 64 key blocks

static __device__ __forceinline__ void stage_kv(const _Float16* __restrict__ Kg,
    const _Float16* __restrict__ Vg, _Float16* Ks, _Float16* Vs, int kb, int tid)
{
  // K tile: 32 rows(key) x 64 f16; 8x16B chunks per row; 256 threads = 1 chunk each
  {
    const int row = tid >> 3, c = (tid & 7) * 8;
    ASYNC_CP16(Ks + row*KS_STRIDE + c, Kg + (size_t)(kb + row)*D_HEAD + c);
  }
  // V tile: 64 rows(d) x 32 f16; 4x16B chunks per row
  {
    const int d = tid >> 2, c = (tid & 3) * 8;
    ASYNC_CP16(Vs + d*VS_STRIDE + c, Vg + (size_t)d*SEQ + kb + c);
  }
}

__global__ __launch_bounds__(256) void flash_kernel(const _Float16* __restrict__ Q,
    const _Float16* __restrict__ K, const _Float16* __restrict__ Vt,
    _Float16* __restrict__ O)
{
  __shared__ __align__(16) _Float16 Ks[2][32*KS_STRIDE];
  __shared__ __align__(16) _Float16 Vs[2][64*VS_STRIDE];
  __shared__ __align__(16) _Float16 plds[8][16*KS_STRIDE];
  const int tid  = threadIdx.x;
  const int lane = tid & 31;
  const int wv   = tid >> 5;
  const int bh   = blockIdx.y;                 // 0..31
  const int q0   = blockIdx.x*128 + wv*16;     // first query row of this wave
  const int half = lane >> 4;
  const float SCL = 0.125f;                    // 1/sqrt(64)
  const float L2E = 1.44269504088896f;

  const _Float16* Qb = Q  + ((size_t)bh*SEQ + q0) * D_HEAD;
  const _Float16* Kb = K  + (size_t)bh*SEQ*D_HEAD;
  const _Float16* Vb = Vt + (size_t)bh*D_HEAD*SEQ;

  // Q fragments are loop-invariant: hoist into registers.
  const v16h aq0 = frag_a(Qb, D_HEAD, 0,  lane);
  const v16h aq1 = frag_a(Qb, D_HEAD, 32, lane);

  v8f acc[4] = {};
  float mrow[8], lrow[8];
  #pragma unroll
  for (int r = 0; r < 8; ++r){ mrow[r] = -1e30f; lrow[r] = 0.0f; }

  _Float16* pw = &plds[wv][0];

  stage_kv(Kb, Vb, Ks[0], Vs[0], 0, tid);      // prologue: stage block 0

  for (int i = 0; i < NKB; ++i){
    const int cur = i & 1;
    if (i + 1 < NKB){
      stage_kv(Kb, Vb, Ks[cur^1], Vs[cur^1], (i+1)*32, tid);
      WAIT_ASYNC(2);                           // block i done, block i+1 in flight
    } else {
      WAIT_ASYNC(0);
    }
    __syncthreads();

    const _Float16* Kc = Ks[cur];
    const _Float16* Vc = Vs[cur];

    // S = Q K^T  (16 queries x 32 keys), K read from LDS
    v8f s0 = {}, s1 = {};
    s0 = WMMA(aq0, frag_b(Kc,               KS_STRIDE, 0,  lane), s0);
    s1 = WMMA(aq0, frag_b(Kc + 16*KS_STRIDE,KS_STRIDE, 0,  lane), s1);
    s0 = WMMA(aq1, frag_b(Kc,               KS_STRIDE, 32, lane), s0);
    s1 = WMMA(aq1, frag_b(Kc + 16*KS_STRIDE,KS_STRIDE, 32, lane), s1);

    // online softmax (row r lives at row r+half*8; cols striped over 16 lanes)
    float p0[8], p1[8];
    #pragma unroll
    for (int r = 0; r < 8; ++r){
      float l0 = s0[r]*SCL, l1 = s1[r]*SCL;
      float mx = fmaxf(l0, l1);
      #pragma unroll
      for (int msk = 1; msk < 16; msk <<= 1) mx = fmaxf(mx, __shfl_xor(mx, msk, 32));
      float mn = fmaxf(mrow[r], mx);
      float al = exp2f((mrow[r] - mn)*L2E);
      p0[r] = exp2f((l0 - mn)*L2E);
      p1[r] = exp2f((l1 - mn)*L2E);
      float rs = p0[r] + p1[r];
      #pragma unroll
      for (int msk = 1; msk < 16; msk <<= 1) rs += __shfl_xor(rs, msk, 32);
      lrow[r] = lrow[r]*al + rs;
      mrow[r] = mn;
      #pragma unroll
      for (int t = 0; t < 4; ++t) acc[t][r] *= al;
    }
    // transpose P through per-wave LDS: C-layout -> A-fragment layout
    {
      const int col = lane & 15;
      #pragma unroll
      for (int r = 0; r < 8; ++r){
        int row = r + half*8;
        pw[row*KS_STRIDE + col]      = (_Float16)p0[r];
        pw[row*KS_STRIDE + col + 16] = (_Float16)p1[r];
      }
    }
    v16h ap = frag_a(pw, KS_STRIDE, 0, lane);  // ds_load_b128 x2
    // O += P V, V read from LDS
    #pragma unroll
    for (int t = 0; t < 4; ++t){
      v16h bv = frag_b(Vc + t*16*VS_STRIDE, VS_STRIDE, 0, lane);
      acc[t] = WMMA(ap, bv, acc[t]);
    }
    __syncthreads();                           // buffer reuse fence
  }

  // epilogue: normalize, store f16 into O[b][s][h*64+d]
  const int b = bh >> 3, h = bh & 7;
  const int coln = lane & 15;
  #pragma unroll
  for (int r = 0; r < 8; ++r){
    float inv = 1.0f / lrow[r];
    int row = q0 + r + half*8;
    size_t ro = ((size_t)b*SEQ + row)*D_MODEL + h*D_HEAD;
    #pragma unroll
    for (int t = 0; t < 4; ++t)
      O[ro + t*16 + coln] = (_Float16)(acc[t][r]*inv);
  }
}

// ---------------- output projection GEMM (fp32 out) ----------------
__global__ __launch_bounds__(256) void outproj_kernel(const _Float16* __restrict__ A,
    const _Float16* __restrict__ WT, const float* __restrict__ bias,
    float* __restrict__ out)
{
  const int lane = threadIdx.x & 31;
  const int w    = threadIdx.x >> 5;
  const int m0 = blockIdx.x*128 + (w & 3)*32;
  const int n0 = blockIdx.y*64  + (w >> 2)*32;
  v8f c[2][2] = {};
  for (int k = 0; k < D_MODEL; k += 32){
    __builtin_prefetch(A  + (size_t)m0*D_MODEL + k + 64, 0, 1);
    v16h a0 = frag_a(A  + (size_t)m0*D_MODEL,      D_MODEL, k, lane);
    v16h a1 = frag_a(A  + (size_t)(m0+16)*D_MODEL, D_MODEL, k, lane);
    v16h b0 = frag_b(WT + (size_t)n0*D_MODEL,      D_MODEL, k, lane);
    v16h b1 = frag_b(WT + (size_t)(n0+16)*D_MODEL, D_MODEL, k, lane);
    c[0][0] = WMMA(a0, b0, c[0][0]);
    c[0][1] = WMMA(a0, b1, c[0][1]);
    c[1][0] = WMMA(a1, b0, c[1][0]);
    c[1][1] = WMMA(a1, b1, c[1][1]);
  }
  const int half = lane >> 4, nn = lane & 15;
  #pragma unroll
  for (int tm = 0; tm < 2; ++tm)
  #pragma unroll
  for (int tn = 0; tn < 2; ++tn)
  #pragma unroll
  for (int r = 0; r < 8; ++r){
    int row = m0 + tm*16 + r + half*8;
    int col = n0 + tn*16 + nn;
    out[(size_t)row*D_MODEL + col] = c[tm][tn][r] + bias[col];
  }
}

extern "C" void kernel_launch(void* const* d_in, const int* in_sizes, int n_in,
                              void* d_out, int out_size, void* d_ws, size_t ws_size,
                              hipStream_t stream)
{
  (void)in_sizes; (void)n_in; (void)out_size; (void)ws_size;
  const float* x  = (const float*)d_in[0];
  const float* Wq = (const float*)d_in[1];
  const float* bq = (const float*)d_in[2];
  const float* Wk = (const float*)d_in[3];
  const float* bk = (const float*)d_in[4];
  const float* Wv = (const float*)d_in[5];
  const float* bv = (const float*)d_in[6];
  const float* Wo = (const float*)d_in[7];
  const float* bo = (const float*)d_in[8];
  float* out = (float*)d_out;

  // workspace layout (f16 elems): total ~44 MB
  _Float16* ws  = (_Float16*)d_ws;
  const size_t NE = (size_t)M_TOTAL * D_MODEL;          // 4194304
  _Float16* xh  = ws;                                   // [8192][512]
  _Float16* WT  = xh + NE;                              // 4 x [512][512] transposed
  _Float16* Qh  = WT + 4u*D_MODEL*D_MODEL;              // [32][2048][64]
  _Float16* Kh  = Qh + NE;                              // [32][2048][64]
  _Float16* Vth = Kh + NE;                              // [32][64][2048]
  _Float16* Oh  = Vth + NE;                             // [8192][512]

  cvt_x_kernel<<<dim3((unsigned)(NE/1024)), 256, 0, stream>>>(x, xh, (int)NE);
  cvt_w_kernel<<<dim3(1024, 4), 256, 0, stream>>>(Wq, Wk, Wv, Wo, WT);

  proj_kernel<<<dim3(64, 8), 256, 0, stream>>>(xh, WT,                                bq, Qh,  0);
  proj_kernel<<<dim3(64, 8), 256, 0, stream>>>(xh, WT +   (size_t)D_MODEL*D_MODEL,    bk, Kh,  1);
  proj_kernel<<<dim3(64, 8), 256, 0, stream>>>(xh, WT + 2*(size_t)D_MODEL*D_MODEL,    bv, Vth, 2);

  flash_kernel<<<dim3(16, 32), 256, 0, stream>>>(Qh, Kh, Vth, Oh);

  outproj_kernel<<<dim3(64, 8), 256, 0, stream>>>(Oh, WT + 3*(size_t)D_MODEL*D_MODEL, bo, out);
}
